// JTGraphEncoder_5927054868744
// MI455X (gfx1250) — matrix operand order repaired and checked by reading, not compiled, not run on hardware
//
#include <hip/hip_runtime.h>

// ---------------------------------------------------------------------------
// Fused JT graph encoder for MI455X (gfx1250, wave32, WMMA bf16).
//   mol = segment_sum( sigmoid(e@U' + x@V' + y@W' + b) * (e@A' + A_b) )
//
// Block = 64 edges (4 row-tiles x 16), 8 waves. Each wave owns 2 column tiles.
// Inner loop streams one B fragment at a time across 4 row-tiles to keep the
// live VGPR set ~90 regs (no spills), while amortizing L2 weight traffic 4x.
// ---------------------------------------------------------------------------

typedef __attribute__((ext_vector_type(16))) __bf16 v16bf;
typedef __attribute__((ext_vector_type(8)))  __bf16 v8bf;
typedef __attribute__((ext_vector_type(8)))  float  v8f;

#define HDIM   256
#define KTILES 8        // 256 / 32  (K per WMMA = 32)
#define RTILES 4        // row tiles per block
#define TILE_E 64       // edges per block = RTILES * 16
#define LDAP   264      // padded LDS row stride (bf16 elems): 528B = 132 dwords

static __device__ __forceinline__ unsigned short f2bf(float f) {
  union { float f; unsigned u; } v; v.f = f;
  unsigned u = v.u;
  u += 0x7fffu + ((u >> 16) & 1u);   // round-to-nearest-even
  return (unsigned short)(u >> 16);
}

// A fragment (16x32 bf16): lane holds row = lane%16; element j<8 -> K = half*8+j,
// element j>=8 -> K = 16 + half*8 + (j-8).  Two 16B LDS chunks, 32B apart.
static __device__ __forceinline__ v16bf load_a_frag(const unsigned short* p) {
  v8bf lo = *(const v8bf*)(p);
  v8bf hi = *(const v8bf*)(p + 16);
  return __builtin_shufflevector(lo, hi, 0,1,2,3,4,5,6,7,8,9,10,11,12,13,14,15);
}

// B fragment: packed contiguously per lane (32B) by pack_weights.
static __device__ __forceinline__ v16bf load_b_frag(const unsigned short* p) {
  v8bf lo = *(const v8bf*)(p);
  v8bf hi = *(const v8bf*)(p + 8);
  return __builtin_shufflevector(lo, hi, 0,1,2,3,4,5,6,7,8,9,10,11,12,13,14,15);
}

// ---------------------------------------------------------------------------
// Repack the four 256x256 f32 weight matrices into bf16 WMMA-B fragment order:
//   packed[mat][ct][kt][lane][j] = W[ct*16 + (lane&15)][kt*32 + 16*(j>>3) + (lane>>4)*8 + (j&7)]
// so B[k][c] = W[c][k] (the @ W.T operand), one contiguous 32B load per lane.
// ---------------------------------------------------------------------------
__global__ void pack_weights(const float* __restrict__ U, const float* __restrict__ V,
                             const float* __restrict__ W, const float* __restrict__ A,
                             unsigned short* __restrict__ packed) {
  int t    = blockIdx.x * blockDim.x + threadIdx.x;   // 0 .. 4*65536-1
  int j    =  t        & 15;
  int lane = (t >> 4)  & 31;
  int kt   = (t >> 9)  & 7;
  int ct   = (t >> 12) & 15;
  int mat  = (t >> 16) & 3;
  const float* src = (mat == 0) ? U : (mat == 1) ? V : (mat == 2) ? W : A;
  int c = ct * 16 + (lane & 15);
  int k = kt * 32 + (j < 8 ? (lane >> 4) * 8 + j : 16 + (lane >> 4) * 8 + (j - 8));
  packed[t] = f2bf(src[c * HDIM + k]);
}

__global__ void bias_sum(const float* __restrict__ Ub, const float* __restrict__ Vb,
                         const float* __restrict__ Wb, float* __restrict__ bsum) {
  int i = threadIdx.x;
  bsum[i] = Ub[i] + Vb[i] + Wb[i];
}

__global__ void zero_out(float* __restrict__ out, int n) {
  int i = blockIdx.x * blockDim.x + threadIdx.x;
  if (i < n) out[i] = 0.0f;
}

// ---------------------------------------------------------------------------
// Fused gather + 4xGEMM + sigmoid-gate + segment-sum.
// ---------------------------------------------------------------------------
__global__ __launch_bounds__(256, 2)
void fused_encoder(const float* __restrict__ edge_feats,
                   const float* __restrict__ node_feats,
                   const int*   __restrict__ edge_node_idx,   // (E,2)
                   const int*   __restrict__ segment_ids,     // (E,) sorted
                   const unsigned short* __restrict__ packed, // B fragments
                   const float* __restrict__ bsum,            // U_b+V_b+W_b
                   const float* __restrict__ A_b,
                   float*       __restrict__ out)             // (M,256), pre-zeroed
{
  __shared__ unsigned short sE[TILE_E * LDAP];
  __shared__ unsigned short sX[TILE_E * LDAP];
  __shared__ unsigned short sY[TILE_E * LDAP];

  const int  tid  = threadIdx.x;
  const long base = (long)blockIdx.x * TILE_E;

  // ---- cooperative gather of e / x / y rows, f32 -> bf16 into LDS ----
  {
    const int r0 = tid >> 4;       // first row (0..15), 16 rows per pass
    const int cg = tid & 15;       // 16 threads per row
#pragma unroll
    for (int rr = 0; rr < TILE_E / 16; ++rr) {
      const int  r = r0 + rr * 16;
      const long e = base + r;
      const float* eRow = edge_feats + (size_t)e * HDIM;
      const int xi = edge_node_idx[2 * e + 0];
      const int yi = edge_node_idx[2 * e + 1];
      const float* xRow = node_feats + (size_t)xi * HDIM;
      const float* yRow = node_feats + (size_t)yi * HDIM;
#pragma unroll
      for (int p = 0; p < 4; ++p) {
        const int c = cg * 4 + p * 64;
        float4 ev = *(const float4*)(eRow + c);
        float4 xv = *(const float4*)(xRow + c);
        float4 yv = *(const float4*)(yRow + c);
        unsigned short* de = &sE[r * LDAP + c];
        unsigned short* dx = &sX[r * LDAP + c];
        unsigned short* dy = &sY[r * LDAP + c];
        de[0]=f2bf(ev.x); de[1]=f2bf(ev.y); de[2]=f2bf(ev.z); de[3]=f2bf(ev.w);
        dx[0]=f2bf(xv.x); dx[1]=f2bf(xv.y); dx[2]=f2bf(xv.z); dx[3]=f2bf(xv.w);
        dy[0]=f2bf(yv.x); dy[1]=f2bf(yv.y); dy[2]=f2bf(yv.z); dy[3]=f2bf(yv.w);
      }
    }
  }
  __syncthreads();

  const int lane = tid & 31;
  const int wave = tid >> 5;
  const int half = lane >> 4;     // 0: rows 0-7 / K-lo, 1: rows 8-15 / K-hi
  const int lrow = lane & 15;     // A row-in-tile / C column-in-tile

#pragma unroll 1
  for (int cti = 0; cti < 2; ++cti) {
    const int ct = wave * 2 + cti;
    v8f accS[RTILES] = {};     // synaptic accumulators (U,V,W streams)
    v8f accA[RTILES] = {};     // gate-value accumulator (A stream)

#pragma unroll 2
    for (int kt = 0; kt < KTILES; ++kt) {
      const size_t toff = ((size_t)(ct * KTILES + kt)) * 512 + lane * 16;
      const int    aoff = kt * 32 + half * 8;

      // ---- U stream: b reused across 4 row tiles, then retired ----
      {
        const v16bf b = load_b_frag(packed + 0 * 65536 + toff);
#pragma unroll
        for (int rt = 0; rt < RTILES; ++rt) {
          v16bf a = load_a_frag(&sE[(rt * 16 + lrow) * LDAP + aoff]);
          accS[rt] = __builtin_amdgcn_wmma_f32_16x16x32_bf16(false, a, false, b,
                                                             (short)0, accS[rt], false, false);
        }
      }
      // ---- V stream (x = node_feats[idx0]) ----
      {
        const v16bf b = load_b_frag(packed + 1 * 65536 + toff);
#pragma unroll
        for (int rt = 0; rt < RTILES; ++rt) {
          v16bf a = load_a_frag(&sX[(rt * 16 + lrow) * LDAP + aoff]);
          accS[rt] = __builtin_amdgcn_wmma_f32_16x16x32_bf16(false, a, false, b,
                                                             (short)0, accS[rt], false, false);
        }
      }
      // ---- W stream (y = node_feats[idx1]) ----
      {
        const v16bf b = load_b_frag(packed + 2 * 65536 + toff);
#pragma unroll
        for (int rt = 0; rt < RTILES; ++rt) {
          v16bf a = load_a_frag(&sY[(rt * 16 + lrow) * LDAP + aoff]);
          accS[rt] = __builtin_amdgcn_wmma_f32_16x16x32_bf16(false, a, false, b,
                                                             (short)0, accS[rt], false, false);
        }
      }
      // ---- A stream (gate value, edge feats again) ----
      {
        const v16bf b = load_b_frag(packed + 3 * 65536 + toff);
#pragma unroll
        for (int rt = 0; rt < RTILES; ++rt) {
          v16bf a = load_a_frag(&sE[(rt * 16 + lrow) * LDAP + aoff]);
          accA[rt] = __builtin_amdgcn_wmma_f32_16x16x32_bf16(false, a, false, b,
                                                             (short)0, accA[rt], false, false);
        }
      }
    }

    // ---- epilogue: bias + sigmoid gate + gated value + segment-sum ----
    const int col = ct * 16 + lrow;
    const float bs = bsum[col];
    const float ab = A_b[col];

#pragma unroll
    for (int rt = 0; rt < RTILES; ++rt) {
      const long rbase = base + rt * 16;
      float g[8];
#pragma unroll
      for (int i = 0; i < 8; ++i) {
        const float syn  = accS[rt][i] + bs;
        const float gate = 1.0f / (1.0f + __expf(-syn));
        g[i] = gate * (accA[rt][i] + ab);
      }

      const int seg0 = segment_ids[rbase];
      if (segment_ids[rbase + 15] == seg0) {          // sorted -> whole tile uniform
        float p = 0.0f;
#pragma unroll
        for (int i = 0; i < 8; ++i) p += g[i];
        p += __shfl_xor(p, 16, 32);                   // combine rows 0-7 with 8-15
        if (half == 0)
          atomicAdd(out + (size_t)seg0 * HDIM + col, p);
      } else {
#pragma unroll
        for (int i = 0; i < 8; ++i) {
          const int s = segment_ids[rbase + i + half * 8];
          atomicAdd(out + (size_t)s * HDIM + col, g[i]);
        }
      }
    }
  }
}

// ---------------------------------------------------------------------------
extern "C" void kernel_launch(void* const* d_in, const int* in_sizes, int n_in,
                              void* d_out, int out_size, void* d_ws, size_t ws_size,
                              hipStream_t stream) {
  const float* node_feats = (const float*)d_in[0];
  const float* edge_feats = (const float*)d_in[1];
  const float* U_w = (const float*)d_in[2];
  const float* U_b = (const float*)d_in[3];
  const float* V_w = (const float*)d_in[4];
  const float* V_b = (const float*)d_in[5];
  const float* W_w = (const float*)d_in[6];
  const float* W_b = (const float*)d_in[7];
  const float* A_w = (const float*)d_in[8];
  const float* A_b = (const float*)d_in[9];
  const int* edge_node_idx = (const int*)d_in[10];
  const int* segment_ids   = (const int*)d_in[11];

  const int E = in_sizes[1] / HDIM;            // 262144

  unsigned short* packed = (unsigned short*)d_ws;                      // 512 KB
  float* bsum = (float*)((char*)d_ws + 4u * HDIM * HDIM * sizeof(unsigned short));

  pack_weights<<<(4 * HDIM * HDIM) / 256, 256, 0, stream>>>(U_w, V_w, W_w, A_w, packed);
  bias_sum<<<1, HDIM, 0, stream>>>(U_b, V_b, W_b, bsum);
  zero_out<<<(out_size + 255) / 256, 256, 0, stream>>>((float*)d_out, out_size);

  fused_encoder<<<E / TILE_E, 256, 0, stream>>>(
      edge_feats, node_feats, edge_node_idx, segment_ids,
      packed, bsum, A_b, (float*)d_out);
}
